// FlatMemoryCell_3642132267553
// MI455X (gfx1250) — matrix operand order repaired and checked by reading, not compiled, not run on hardware
//
#include <hip/hip_runtime.h>
#include <hip/hip_bf16.h>

// ---------------------------------------------------------------------------
// FlatMemoryCell on MI455X (gfx1250, wave32, WMMA)
//   GEMM1: [B,1792]x[1792,1024] bf16 WMMA, fp32 accum, reg-double-buffered
//   GEMM2: [B,1280]x[1280,144]  bf16 WMMA, async global->LDS double-buffered
//   elementwise: factorized p5 norm + gated outer-product memory update
// Ext-vector types force b128 global loads; 32-bit offsets enable SADDR forms.
// ---------------------------------------------------------------------------

#define BDIM   16384
#define IND    512
#define HIDD   1024
#define MEMD   256
#define KIN1   1792            // IN + HID + MEM
#define KIN2   1280            // HID + MEM
#define NPAD2  144             // 136 outputs padded to 9 WMMA tiles
#define LDT    40              // LDS row stride (shorts): 80B = 16B-aligned rows,
                               // 20*m mod 64 permutation -> conflict-free frag rows

typedef __attribute__((ext_vector_type(16))) __bf16 v16bf;
typedef __attribute__((ext_vector_type(8)))  float  v8f;
typedef __attribute__((ext_vector_type(4)))  float  v4f;
typedef __attribute__((ext_vector_type(4)))  int    v4i;

struct U8 { unsigned u[8]; };

#define AS1 __attribute__((address_space(1)))
#define AS3 __attribute__((address_space(3)))

#if __has_builtin(__builtin_amdgcn_global_load_async_to_lds_b128) && \
    __has_builtin(__builtin_amdgcn_s_wait_asynccnt)
#define HAVE_ASYNC_LDS 1
#else
#define HAVE_ASYNC_LDS 0
#endif

static __device__ __forceinline__ unsigned short f2bf(float f) {
    unsigned u = __float_as_uint(f);
    return (unsigned short)((u + 0x7FFFu + ((u >> 16) & 1u)) >> 16);
}

static __device__ __forceinline__ unsigned pack2bf(float a, float b) {
#if __has_builtin(__builtin_amdgcn_cvt_pk_bf16_f32)
    auto r = __builtin_amdgcn_cvt_pk_bf16_f32(a, b);   // v_cvt_pk_bf16_f32
    return __builtin_bit_cast(unsigned, r);
#else
    return (unsigned)f2bf(a) | ((unsigned)f2bf(b) << 16);
#endif
}

// 16B global -> LDS copy; async path (ASYNCcnt) when available
static __device__ __forceinline__ void cp16_g2l(const unsigned short* g, unsigned short* l) {
#if HAVE_ASYNC_LDS
    __builtin_amdgcn_global_load_async_to_lds_b128(
        (AS1 v4i*)(unsigned long long)(const void*)g, (AS3 v4i*)(void*)l, 0, 0);
#else
    *(v4i*)(void*)l = *(const v4i*)g;                 // b128 load + ds_store_b128
#endif
}
static __device__ __forceinline__ void wait_async_all() {
#if HAVE_ASYNC_LDS
    __builtin_amdgcn_s_wait_asynccnt(0);
#endif
}

// A-operand fragment (16x32 bf16) in contiguous k order (same as frag_b):
// lane L (m=L&15, kh=L>>4): VGPR v: K = 16*kh + 2v
static __device__ __forceinline__ v16bf frag_a(const unsigned short* lds, int rowBase, int lane) {
    const unsigned* p = (const unsigned*)(lds + (rowBase + (lane & 15)) * LDT);
    int kh4 = (lane >> 4) << 3;
    U8 t;
#pragma unroll
    for (int j = 0; j < 4; ++j) t.u[j]     = p[kh4 + j];
#pragma unroll
    for (int j = 0; j < 4; ++j) t.u[4 + j] = p[4 + kh4 + j];
    return __builtin_bit_cast(v16bf, t);
}

// B-operand fragment (32x16 bf16; LDS holds W rows n with K contiguous):
// lane L (n=L&15, kh=L>>4): VGPR v: K = 16*kh + 2v
static __device__ __forceinline__ v16bf frag_b(const unsigned short* lds, int rowBase, int lane) {
    const unsigned* p = (const unsigned*)(lds + (rowBase + (lane & 15)) * LDT) + ((lane >> 4) << 3);
    U8 t;
#pragma unroll
    for (int v = 0; v < 8; ++v) t.u[v] = p[v];
    return __builtin_bit_cast(v16bf, t);
}

// ---------------------------------------------------------------------------
// prep: W2cat bf16 [144x1280] = {W_alpha(4), W_beta(4), W_valpha(64),
//       W_vbeta(64), zeros(8)} + fused bias bcat[144]
// ---------------------------------------------------------------------------
__global__ void k_prep(const float* __restrict__ Wa, const float* __restrict__ ba,
                       const float* __restrict__ Wb, const float* __restrict__ bb,
                       const float* __restrict__ Wva, const float* __restrict__ bva,
                       const float* __restrict__ Wvb, const float* __restrict__ bvb,
                       unsigned short* __restrict__ w2, float* __restrict__ bcat) {
    int i = blockIdx.x * 256 + threadIdx.x;
    if (i < NPAD2 * KIN2) {
        int row = i / KIN2, c = i - row * KIN2;
        float v = 0.f;
        if      (row <   4) v = Wa [ row       * KIN2 + c];
        else if (row <   8) v = Wb [(row -  4) * KIN2 + c];
        else if (row <  72) v = Wva[(row -  8) * KIN2 + c];
        else if (row < 136) v = Wvb[(row - 72) * KIN2 + c];
        w2[i] = f2bf(v);
    }
    if (i < NPAD2) {
        float b = 0.f;
        if      (i <   4) b = ba [i];
        else if (i <   8) b = bb [i - 4];
        else if (i <  72) b = bva[i - 8];
        else if (i < 136) b = bvb[i - 72];
        bcat[i] = b;
    }
}

__global__ void k_memconv(const float* __restrict__ mem, unsigned short* __restrict__ hm) {
    int i = blockIdx.x * 256 + threadIdx.x;          // i < B*256
    int row = i >> 8, c = i & 255;
    hm[row * KIN2 + HIDD + c] = f2bf(mem[i]);
}

// ---------------------------------------------------------------------------
// GEMM1: h_new = relu(concat(x,h,mem) @ W1^T + b1)
//   block tile 256x128, 8 waves, 64x64 per wave (4x4 WMMA tiles)
//   register-prefetch double buffering (fp32->bf16 conversion in stash)
// ---------------------------------------------------------------------------
__global__ void __launch_bounds__(256)
k_gemm1(const float* __restrict__ x, const float* __restrict__ h,
        const float* __restrict__ mem, const float* __restrict__ W1,
        const float* __restrict__ b1, float* __restrict__ hnew_out,
        unsigned short* __restrict__ hm) {
    __shared__ unsigned short As[2][256 * LDT];
    __shared__ unsigned short Bs[2][128 * LDT];

    const int tid   = threadIdx.x;
    const int lane  = tid & 31;
    const int wave  = tid >> 5;
    const int waveM = wave & 3;          // 4 waves along M (64 rows each)
    const int waveN = wave >> 2;         // 2 waves along N (64 cols each)
    const int mBase = blockIdx.y * 256;
    const int nBase = blockIdx.x * 128;

    // per-thread 32-bit base offsets, hoisted out of the k-loop
    const int rowT = tid >> 3, cgT = tid & 7;
    const int offX = (mBase + rowT) * IND  + cgT * 4;
    const int offH = (mBase + rowT) * HIDD + cgT * 4;
    const int offM = (mBase + rowT) * MEMD + cgT * 4;
    const int offW = (nBase + rowT) * KIN1 + cgT * 4;
    const int offL = rowT * LDT + cgT * 4;           // LDS stash base (shorts)

    v4f ra[8], rw[4];

    auto loadA = [&](int kt) {
        const int k0 = kt * 32;
        if (k0 < IND) {
#pragma unroll
            for (int p = 0; p < 8; ++p)
                ra[p] = *(const v4f*)(x + offX + p * (32 * IND) + k0);
        } else if (k0 < IND + HIDD) {
#pragma unroll
            for (int p = 0; p < 8; ++p)
                ra[p] = *(const v4f*)(h + offH + p * (32 * HIDD) + (k0 - IND));
        } else {
#pragma unroll
            for (int p = 0; p < 8; ++p)
                ra[p] = *(const v4f*)(mem + offM + p * (32 * MEMD) + (k0 - IND - HIDD));
        }
    };
    auto loadW = [&](int kt) {
        const int k0 = kt * 32;
#pragma unroll
        for (int p = 0; p < 4; ++p)
            rw[p] = *(const v4f*)(W1 + offW + p * (32 * KIN1) + k0);
    };
    auto stash = [&](int b) {
#pragma unroll
        for (int p = 0; p < 8; ++p) {
            unsigned* d = (unsigned*)&As[b][offL + p * (32 * LDT)];
            d[0] = pack2bf(ra[p].x, ra[p].y); d[1] = pack2bf(ra[p].z, ra[p].w);
        }
#pragma unroll
        for (int p = 0; p < 4; ++p) {
            unsigned* d = (unsigned*)&Bs[b][offL + p * (32 * LDT)];
            d[0] = pack2bf(rw[p].x, rw[p].y); d[1] = pack2bf(rw[p].z, rw[p].w);
        }
    };

    v8f acc[4][4];
#pragma unroll
    for (int i = 0; i < 4; ++i)
#pragma unroll
        for (int j = 0; j < 4; ++j) acc[i][j] = (v8f){0.f,0.f,0.f,0.f,0.f,0.f,0.f,0.f};

    loadA(0); loadW(0);
    for (int kt = 0; kt < KIN1 / 32; ++kt) {
        const int b = kt & 1;
        stash(b);
        __syncthreads();                 // single barrier/iter: 2 LDS buffers
        if (kt + 1 < KIN1 / 32) { loadA(kt + 1); loadW(kt + 1); }  // overlap HBM with WMMA

        v16bf af[4], bv[4];
#pragma unroll
        for (int i = 0; i < 4; ++i) af[i] = frag_a(&As[b][0], waveM * 64 + 16 * i, lane);
#pragma unroll
        for (int j = 0; j < 4; ++j) bv[j] = frag_b(&Bs[b][0], waveN * 64 + 16 * j, lane);
#pragma unroll
        for (int i = 0; i < 4; ++i)
#pragma unroll
            for (int j = 0; j < 4; ++j)
                acc[i][j] = __builtin_amdgcn_wmma_f32_16x16x32_bf16(
                    false, af[i], false, bv[j], (short)0, acc[i][j], false, false);
    }

    const int mo = 8 * (lane >> 4), no = lane & 15;
#pragma unroll
    for (int i = 0; i < 4; ++i)
#pragma unroll
        for (int j = 0; j < 4; ++j) {
            int col = nBase + waveN * 64 + 16 * j + no;
            float bias = b1[col];
#pragma unroll
            for (int r = 0; r < 8; ++r) {
                int row = mBase + waveM * 64 + 16 * i + mo + r;
                float v = fmaxf(acc[i][j][r] + bias, 0.f);
                hnew_out[row * HIDD + col] = v;
                hm[row * KIN2 + col] = f2bf(v);
            }
        }
}

// ---------------------------------------------------------------------------
// GEMM2: u = hm @ W2cat^T + bcat -> U[B,144]
//   block tile 256x144, 8 waves, 32 rows x 144 cols per wave (2x9 tiles)
//   async global->LDS double buffering (ASYNCcnt) when available
// ---------------------------------------------------------------------------
__global__ void __launch_bounds__(256)
k_gemm2(const unsigned short* __restrict__ hm, const unsigned short* __restrict__ w2,
        const float* __restrict__ bcat, float* __restrict__ U) {
    __shared__ unsigned short As[2][256 * LDT];
    __shared__ unsigned short Bs[2][NPAD2 * LDT];

    const int tid  = threadIdx.x;
    const int lane = tid & 31;
    const int wave = tid >> 5;
    const int mBase = blockIdx.x * 256;

    const int rowT = tid >> 2, qT = tid & 3;         // 64 rows per 256-thread pass
    const int offA = (mBase + rowT) * KIN2 + qT * 8; // shorts
    const int offB = rowT * KIN2 + qT * 8;           // shorts
    const int offL = rowT * LDT + qT * 8;            // shorts

    auto issue = [&](int kt, int b) {
        const int k0 = kt * 32;
#pragma unroll
        for (int p = 0; p < 4; ++p)                  // A tile 256x32 bf16
            cp16_g2l(hm + offA + p * (64 * KIN2) + k0,
                     &As[b][offL + p * (64 * LDT)]);
#pragma unroll
        for (int p = 0; p < 3; ++p) {                // W2 tile 144x32 bf16
            if (p * 64 + rowT < NPAD2)
                cp16_g2l(w2 + offB + p * (64 * KIN2) + k0,
                         &Bs[b][offL + p * (64 * LDT)]);
        }
    };

    v8f acc0[9], acc1[9];
#pragma unroll
    for (int j = 0; j < 9; ++j) {
        acc0[j] = (v8f){0.f,0.f,0.f,0.f,0.f,0.f,0.f,0.f};
        acc1[j] = (v8f){0.f,0.f,0.f,0.f,0.f,0.f,0.f,0.f};
    }

    issue(0, 0);
    wait_async_all();
    __syncthreads();

    for (int kt = 0; kt < KIN2 / 32; ++kt) {
        const int b = kt & 1;
        if (kt + 1 < KIN2 / 32) issue(kt + 1, 1 - b);   // overlaps WMMA below

        v16bf a0 = frag_a(&As[b][0], wave * 32,      lane);
        v16bf a1 = frag_a(&As[b][0], wave * 32 + 16, lane);
#pragma unroll
        for (int j = 0; j < 9; ++j) {
            v16bf bv = frag_b(&Bs[b][0], 16 * j, lane);
            acc0[j] = __builtin_amdgcn_wmma_f32_16x16x32_bf16(
                false, a0, false, bv, (short)0, acc0[j], false, false);
            acc1[j] = __builtin_amdgcn_wmma_f32_16x16x32_bf16(
                false, a1, false, bv, (short)0, acc1[j], false, false);
        }
        wait_async_all();
        __syncthreads();
    }

    const int mo = 8 * (lane >> 4), no = lane & 15;
#pragma unroll
    for (int j = 0; j < 9; ++j) {
        int col = 16 * j + no;
        float bias = bcat[col];
#pragma unroll
        for (int r = 0; r < 8; ++r) {
            int row0 = mBase + wave * 32 + mo + r;
            U[row0 * NPAD2 + col]        = acc0[j][r] + bias;
            U[(row0 + 16) * NPAD2 + col] = acc1[j][r] + bias;
        }
    }
}

// ---------------------------------------------------------------------------
// elementwise: memory_new = memory + 0.25 * sum_k [ a_k*ua0_i*ua1_l/Na_k
//                                                 - b_k*ub0_i*ub1_l/Nb_k ]
//   Na_k = max(||ua0[8k:8k+8]||_5 * ||ua1||_5, 1e-12)  (factorized p5 norm)
//   one wave per batch row
// ---------------------------------------------------------------------------
static __device__ __forceinline__ float pow5abs(float x) {
    float t = fabsf(x); float t2 = t * t; return t2 * t2 * t;
}

__global__ void __launch_bounds__(256)
k_update(const float* __restrict__ U, const float* __restrict__ mem,
         float* __restrict__ out) {
    const int lane = threadIdx.x & 31;
    const int row  = blockIdx.x * 8 + (threadIdx.x >> 5);
    const float* u = U + row * NPAD2;

    float a0 = u[ 8 + lane], a1 = u[ 40 + lane];
    float b0 = u[72 + lane], b1 = u[104 + lane];

    float s0a = pow5abs(a0), s0b = pow5abs(b0);
#pragma unroll
    for (int d = 1; d <= 4; d <<= 1) { s0a += __shfl_xor(s0a, d); s0b += __shfl_xor(s0b, d); }
    float s1a = pow5abs(a1), s1b = pow5abs(b1);
#pragma unroll
    for (int d = 1; d <= 16; d <<= 1) { s1a += __shfl_xor(s1a, d); s1b += __shfl_xor(s1b, d); }

    float n0a = powf(s0a, 0.2f), n0b = powf(s0b, 0.2f);
    float n1a = powf(s1a, 0.2f), n1b = powf(s1b, 0.2f);

    float ia[4], ib[4];
#pragma unroll
    for (int k = 0; k < 4; ++k) {
        float na = __shfl(n0a, 8 * k) * n1a;
        float nb = __shfl(n0b, 8 * k) * n1b;
        ia[k] = 0.25f * u[k]     / fmaxf(na, 1e-12f);
        ib[k] = 0.25f * u[4 + k] / fmaxf(nb, 1e-12f);
    }

#pragma unroll
    for (int t = 0; t < 8; ++t) {
        float sA = 0.f, sB = 0.f;
#pragma unroll
        for (int k = 0; k < 4; ++k) {
            sA += ia[k] * __shfl(a0, 8 * k + t);
            sB += ib[k] * __shfl(b0, 8 * k + t);
        }
        int m = 32 * t + lane;
        out[row * MEMD + m] = mem[row * MEMD + m] + a1 * sA - b1 * sB;
    }
}

// ---------------------------------------------------------------------------
extern "C" void kernel_launch(void* const* d_in, const int* in_sizes, int n_in,
                              void* d_out, int out_size, void* d_ws, size_t ws_size,
                              hipStream_t stream) {
    const float* x    = (const float*)d_in[0];
    const float* h    = (const float*)d_in[1];
    const float* mem  = (const float*)d_in[2];
    const float* W1   = (const float*)d_in[3];
    const float* b1   = (const float*)d_in[4];
    const float* Wa   = (const float*)d_in[5];
    const float* ba   = (const float*)d_in[6];
    const float* Wb   = (const float*)d_in[7];
    const float* bb   = (const float*)d_in[8];
    const float* Wva  = (const float*)d_in[9];
    const float* bva  = (const float*)d_in[10];
    const float* Wvb  = (const float*)d_in[11];
    const float* bvb  = (const float*)d_in[12];

    float* out_mem  = (float*)d_out;                       // [B,256]
    float* out_hnew = (float*)d_out + (size_t)BDIM * MEMD; // [B,1024]

    char* ws = (char*)d_ws;
    unsigned short* hm   = (unsigned short*)ws;                          // B*1280 bf16
    size_t off = (size_t)BDIM * KIN2 * 2;
    unsigned short* w2   = (unsigned short*)(ws + off);  off += (size_t)NPAD2 * KIN2 * 2;
    float*          bcat = (float*)(ws + off);           off += 1024;
    float*          U    = (float*)(ws + off);                           // B*144 fp32

    k_prep<<<(NPAD2 * KIN2 + 255) / 256, 256, 0, stream>>>(Wa, ba, Wb, bb, Wva, bva, Wvb, bvb, w2, bcat);
    k_memconv<<<(BDIM * MEMD) / 256, 256, 0, stream>>>(mem, hm);
    k_gemm1<<<dim3(HIDD / 128, BDIM / 256), 256, 0, stream>>>(x, h, mem, W1, b1, out_hnew, hm);
    k_gemm2<<<BDIM / 256, 256, 0, stream>>>(hm, w2, bcat, U);
    k_update<<<BDIM / 8, 256, 0, stream>>>(U, mem, out_mem);
}
